// DCConv_2886218023584
// MI455X (gfx1250) — 8-bit matrix operand order repaired and checked by reading, compiled, not run on hardware
//
#include <hip/hip_runtime.h>
#include <hip/hip_bf16.h>
#include <stdint.h>

typedef int v8i __attribute__((ext_vector_type(8)));

#define XN 32
#define XC 256
#define XH 56
#define XW 56
#define PH 58
#define PW 58
#define OCN 256
#define SPAT (XH * XW)
#define M_TOTAL (XN * SPAT)
#define M_TILE 128
#define OC_WG 128
#define XU8P_BYTES ((size_t)XN * PH * PW * XC)
#define BPANEL_BYTES (9 * 4 * 16 * 32 * 32)

__device__ __forceinline__ void async_ld_b128(uint32_t lds_off, uint32_t voff,
                                              const uint8_t* base) {
    asm volatile("global_load_async_to_lds_b128 %0, %1, %2"
                 :: "v"(lds_off), "v"(voff), "s"(base) : "memory");
}

__global__ void pack_x_kernel(const int* __restrict__ x, uint8_t* __restrict__ xp) {
    const int nh = blockIdx.x;
    const int n = nh / XH, h = nh % XH;
    const int c = threadIdx.x;
    const int4* src = (const int4*)(x + (((n * XC + c) * XH + h) * XW));
    const int pos = (c & 192) + ((c >> 4) & 1) * 32 + ((c >> 5) & 1) * 16 + (c & 15);
    uint8_t* dst = xp + ((size_t)((n * PH + h + 1) * PW + 1)) * XC + pos;
#pragma unroll
    for (int w4 = 0; w4 < XW / 4; ++w4) {
        int4 v = src[w4];
        dst[(w4 * 4 + 0) * XC] = (uint8_t)v.x;
        dst[(w4 * 4 + 1) * XC] = (uint8_t)v.y;
        dst[(w4 * 4 + 2) * XC] = (uint8_t)v.z;
        dst[(w4 * 4 + 3) * XC] = (uint8_t)v.w;
    }
}

__global__ void pack_w_kernel(const int* __restrict__ w, uint8_t* __restrict__ bp) {
    const int t = blockIdx.x * 256 + threadIdx.x;
    const int b  = t & 31;
    const int l  = (t >> 5) & 31;
    const int ot = (t >> 10) & 15;
    const int ch = (t >> 14) & 3;
    const int kw = (t >> 16) % 3;
    const int kh = t / 196608;
    const int v = b >> 2, c = b & 3;
    const int K = ((v >> 2) << 5) + ((l >> 4) << 4) + ((v & 3) << 2) + c;
    const int ic = ch * 64 + K;
    const int oc = ot * 16 + (l & 15);
    bp[t] = (uint8_t)w[((oc * XC + ic) * 3 + kh) * 3 + kw];
}

__global__ __launch_bounds__(256, 1)
void conv_wmma_kernel(const uint8_t* __restrict__ xp,
                      const uint8_t* __restrict__ bp,
                      int* __restrict__ out) {
    __shared__ __align__(16) uint8_t sA[2][M_TILE * 128];
    __shared__ __align__(16) uint8_t sB[2][2 * 8 * 1024];

    const int tid  = threadIdx.x;
    const int lane = tid & 31;
    const int wv   = tid >> 5;
    const int wm   = wv & 3;
    const int wn   = wv >> 2;
    const int m_base = blockIdx.x * M_TILE;
    const int ocg = blockIdx.y;

    const int r    = tid >> 1;
    const int part = tid & 1;
    const int m  = m_base + r;
    const int n  = m / SPAT;
    const int rm = m - n * SPAT;
    const int oh = rm / XW;
    const int ow = rm - oh * XW;
    const uint32_t a_g0 = (uint32_t)(((n * PH + oh) * PW + ow) * XC + part * 64);
    const int sb_ch2 = tid >> 7;
    const int sb_nt  = (tid >> 4) & 7;
    const int sb_in  = (tid & 15) << 6;
    const uint32_t b_g0 = (uint32_t)((sb_ch2 << 14) + ((ocg * 8 + sb_nt) << 10) + sb_in);

    const uint32_t sA_l0 = (uint32_t)(uintptr_t)(&sA[0][0]) +
                           (uint32_t)((r << 7) + part * 64);
    const uint32_t sB_l0 = (uint32_t)(uintptr_t)(&sB[0][0]) +
                           (uint32_t)((sb_ch2 << 13) + (sb_nt << 10) + sb_in);

    auto issue = [&](int s, int buf) {
        const int it = s >> 1, hs = s & 1;
        const int kh = it / 3, kw = it - kh * 3;
        const uint32_t ag = a_g0 + (uint32_t)((kh * PW + kw) * XC + hs * 128);
        const uint32_t al = sA_l0 + (uint32_t)(buf << 14);
        const uint32_t bg = b_g0 + (uint32_t)((it * 4 + hs * 2) << 14);
        const uint32_t bl = sB_l0 + (uint32_t)(buf << 14);
#pragma unroll
        for (int i = 0; i < 4; ++i) async_ld_b128(al + i * 16, ag + i * 16, xp);
#pragma unroll
        for (int i = 0; i < 4; ++i) async_ld_b128(bl + i * 16, bg + i * 16, bp);
    };

    v8i acc[2][4];
#pragma unroll
    for (int am = 0; am < 2; ++am)
#pragma unroll
        for (int j = 0; j < 4; ++j) acc[am][j] = v8i{0, 0, 0, 0, 0, 0, 0, 0};

    const int arow0 = wm * 32 + (lane & 15);
    const int agrp  = (lane >= 16) ? 32 : 0;

    issue(0, 0);
    for (int s = 0; s < 18; ++s) {
        const int buf = s & 1;
        if (s < 17) {
            issue(s + 1, buf ^ 1);
            asm volatile("s_wait_asynccnt 0x8" ::: "memory");
        } else {
            asm volatile("s_wait_asynccnt 0x0" ::: "memory");
        }
        __syncthreads();

        const uint8_t* sAb = &sA[buf][0];
        const uint8_t* sBb = &sB[buf][0];
#pragma unroll
        for (int ch2 = 0; ch2 < 2; ++ch2) {
            v8i a[2];
#pragma unroll
            for (int am = 0; am < 2; ++am) {
                const uint8_t* ap = sAb + ((arow0 + am * 16) << 7) + (ch2 << 6) + agrp;
                int4 p0 = *(const int4*)(ap);
                int4 p1 = *(const int4*)(ap + 16);
                a[am] = v8i{p0.x, p0.y, p0.z, p0.w, p1.x, p1.y, p1.z, p1.w};
            }
            const uint8_t* bbase = sBb + (ch2 << 13) + (wn << 12) + (lane << 5);
#pragma unroll
            for (int j = 0; j < 4; ++j) {
                const uint8_t* bpn = bbase + (j << 10);
                int4 q0 = *(const int4*)(bpn);
                int4 q1 = *(const int4*)(bpn + 16);
                v8i b = v8i{q0.x, q0.y, q0.z, q0.w, q1.x, q1.y, q1.z, q1.w};
                acc[0][j] = __builtin_amdgcn_wmma_i32_16x16x64_iu8(
                    false, a[0], false, b, acc[0][j], false, false);
                acc[1][j] = __builtin_amdgcn_wmma_i32_16x16x64_iu8(
                    false, a[1], false, b, acc[1][j], false, false);
            }
        }
        __syncthreads();
    }

    const int n0  = m_base / SPAT;
    const int bnd = (n0 + 1) * SPAT;
    const int oc0 = ocg * OC_WG + wn * 64 + (lane & 15);
#pragma unroll
    for (int am = 0; am < 2; ++am) {
        const int mrow0 = m_base + wm * 32 + am * 16 + ((lane >= 16) ? 8 : 0);
#pragma unroll
        for (int j = 0; j < 4; ++j) {
            const int oc = oc0 + j * 16;
#pragma unroll
            for (int jj = 0; jj < 8; ++jj) {
                const int mm = mrow0 + jj;
                const int nn = (mm >= bnd) ? (n0 + 1) : n0;
                const int rr = mm - nn * SPAT;
                out[(size_t)(nn * OCN + oc) * SPAT + rr] = acc[am][j][jj];
            }
        }
    }
}

extern "C" void kernel_launch(void* const* d_in, const int* in_sizes, int n_in,
                              void* d_out, int out_size, void* d_ws, size_t ws_size,
                              hipStream_t stream) {
    const int* x = (const int*)d_in[0];
    const int* w = (const int*)d_in[1];
    int* out = (int*)d_out;

    uint8_t* xp  = (uint8_t*)d_ws;
    uint8_t* bpk = xp + XU8P_BYTES;

    hipMemsetAsync(xp, 0, XU8P_BYTES, stream);
    pack_x_kernel<<<XN * XH, 256, 0, stream>>>(x, xp);
    pack_w_kernel<<<BPANEL_BYTES / 256, 256, 0, stream>>>(w, bpk);

    dim3 grid(M_TOTAL / M_TILE, OCN / OC_WG);
    conv_wmma_kernel<<<grid, 256, 0, stream>>>(xp, bpk, out);
}
